// TransformerBlock_31250182046130
// MI455X (gfx1250) — matrix operand order repaired and checked by reading, not compiled, not run on hardware
//
#include <hip/hip_runtime.h>
#include <hip/hip_bf16.h>
#include <math.h>

typedef _Float16 half_t;
typedef __attribute__((ext_vector_type(8)))  _Float16 v8h;
typedef __attribute__((ext_vector_type(16))) _Float16 v16h;
typedef __attribute__((ext_vector_type(8)))  float    v8f;

#define EMBED    1024
#define HEADS    16
#define HEAD_DIM 64
#define HIDDEN   4096
#define SEQ      2048
#define BATCH    2
#define TOKENS   (BATCH*SEQ)
#define EPSV     1e-5f
#define LDP      40   // LDS tile stride in halves: 80B = 16B-aligned, bank-conflict-free

static __device__ __forceinline__ v8f wmma_f16(v16h a, v16h b, v8f c) {
  // D = A(16x32 f16) x B(32x16 f16) + C(16x16 f32)
  return __builtin_amdgcn_wmma_f32_16x16x32_f16(false, a, false, b, (short)0, c, false, false);
}

static __device__ __forceinline__ v16h cat8(v8h lo, v8h hi) {
  return __builtin_shufflevector(lo, hi, 0, 1, 2, 3, 4, 5, 6, 7,
                                 8, 9, 10, 11, 12, 13, 14, 15);
}

// Intra-wave LDS RAW fence: DS ops are tracked per-wave with DScnt (CDNA5 §7.1),
// so waiting DScnt==0 makes same-wave cross-lane LDS writes visible to reads.
static __device__ __forceinline__ void wave_lds_fence() {
  __builtin_amdgcn_wave_barrier();
  asm volatile("s_wait_dscnt 0x0" ::: "memory");
  __builtin_amdgcn_wave_barrier();
}

// ---------------------------------------------------------------------------
// fp32 -> fp16 conversion (grid-stride)
// ---------------------------------------------------------------------------
__global__ void cvt_f32_to_f16(const float* __restrict__ src, half_t* __restrict__ dst, int n) {
  int i = blockIdx.x * blockDim.x + threadIdx.x;
  int stride = gridDim.x * blockDim.x;
  for (; i < n; i += stride) dst[i] = (half_t)src[i];
}

// ---------------------------------------------------------------------------
// LayerNorm (one row of EMBED=1024 per 256-thread block), f32 in -> f16 out
// ---------------------------------------------------------------------------
__global__ void layernorm_f16(const float* __restrict__ x, const float* __restrict__ scale,
                              const float* __restrict__ shift, half_t* __restrict__ out) {
  const int row = blockIdx.x;
  const int tid = threadIdx.x;
  const float* xr = x + (size_t)row * EMBED;
  float vals[4];
  float s = 0.f, ss = 0.f;
#pragma unroll
  for (int j = 0; j < 4; ++j) {
    float v = xr[tid + j * 256];
    vals[j] = v; s += v; ss += v * v;
  }
  __shared__ float r1[256], r2[256];
  r1[tid] = s; r2[tid] = ss;
  __syncthreads();
  for (int off = 128; off > 0; off >>= 1) {
    if (tid < off) { r1[tid] += r1[tid + off]; r2[tid] += r2[tid + off]; }
    __syncthreads();
  }
  const float mean = r1[0] * (1.f / EMBED);
  const float var  = r2[0] * (1.f / EMBED) - mean * mean;
  const float rstd = rsqrtf(var + EPSV);
#pragma unroll
  for (int j = 0; j < 4; ++j) {
    int c = tid + j * 256;
    out[(size_t)row * EMBED + c] = (half_t)(scale[c] * (vals[j] - mean) * rstd + shift[c]);
  }
}

// ---------------------------------------------------------------------------
// WMMA GEMM: C[M,N] = A[M,K](f16) * B[K,N](f16) + bias, optional exact GELU,
// optional f32 residual add. 128x128 block tile, 8 wave32 waves (4 along M,
// 2 along N), each wave 32x64 (2x4 WMMA tiles), K-step 32.
// A tile kept row-major [m][k]; B tile stored TRANSPOSED [n][k] so fragment
// reads are contiguous 16B v8h LDS loads (ds_load_b128).
// ---------------------------------------------------------------------------
template <bool GELU, bool RES>
__global__ void gemm_wmma(const half_t* __restrict__ A, const half_t* __restrict__ Bw,
                          const float* __restrict__ bias, const float* __restrict__ residual,
                          float* __restrict__ outF, half_t* __restrict__ outH,
                          int M, int N, int K) {
  __shared__ __attribute__((aligned(16))) half_t Al[128 * LDP];  // [m][k]
  __shared__ __attribute__((aligned(16))) half_t Bt[128 * LDP];  // [n][k] (transposed)

  const int tid  = threadIdx.x;
  const int lane = tid & 31, wave = tid >> 5;
  const int hi = lane >> 4, ln = lane & 15;
  const int wm = wave & 3, wn = wave >> 2;
  const int bM = blockIdx.y * 128, bN = blockIdx.x * 128;

  v8f acc[2][4];
#pragma unroll
  for (int i = 0; i < 2; i++)
#pragma unroll
    for (int j = 0; j < 4; j++)
#pragma unroll
      for (int e = 0; e < 8; e++) acc[i][j][e] = 0.f;

  const int arow = tid >> 1, ak0 = (tid & 1) * 16;   // 256 thr x 16 halves = 128x32
  const int brow = tid >> 3, bn0 = (tid & 7) * 16;   // 256 thr x 16 halves = 32x128

  for (int kb = 0; kb < K; kb += 32) {
    __syncthreads();
    {
      // A tile: two aligned v8h global loads -> two aligned v8h LDS stores
      const v8h* ap = (const v8h*)(A + (size_t)(bM + arow) * K + kb + ak0);
      v8h* as = (v8h*)(Al + arow * LDP + ak0);
      as[0] = ap[0];
      as[1] = ap[1];
      // B tile: contiguous 16-half global row chunk, scatter transposed to Bt[n][k]
      const half_t* bp = Bw + (size_t)(kb + brow) * N + bN + bn0;
#pragma unroll
      for (int j = 0; j < 16; j++) Bt[(bn0 + j) * LDP + brow] = bp[j];
      if (kb + 32 < K) {
        __builtin_prefetch(A + (size_t)(bM + arow) * K + kb + 32 + ak0, 0, 1);
        __builtin_prefetch(Bw + (size_t)(kb + 32 + brow) * N + bN + bn0, 0, 1);
      }
    }
    __syncthreads();

    v16h af[2], bf[4];
#pragma unroll
    for (int mt = 0; mt < 2; ++mt) {
      // 16-bit A 16x32 layout: element i -> k = 16*hi + i (matches B)
      const half_t* base = Al + (wm * 32 + mt * 16 + ln) * LDP;
      af[mt] = cat8(*(const v8h*)(base + 16 * hi), *(const v8h*)(base + 16 * hi + 8));
    }
#pragma unroll
    for (int nt = 0; nt < 4; ++nt) {
      // 16-bit B 32x16 fragment (this code's k order, same as A): element i -> k = 16*hi + i  (contiguous in Bt[n][k])
      const half_t* base = Bt + (wn * 64 + nt * 16 + ln) * LDP + 16 * hi;
      bf[nt] = cat8(*(const v8h*)(base), *(const v8h*)(base + 8));
    }
#pragma unroll
    for (int mt = 0; mt < 2; ++mt)
#pragma unroll
      for (int nt = 0; nt < 4; ++nt)
        acc[mt][nt] = wmma_f16(af[mt], bf[nt], acc[mt][nt]);
  }

#pragma unroll
  for (int mt = 0; mt < 2; ++mt)
#pragma unroll
    for (int nt = 0; nt < 4; ++nt)
#pragma unroll
      for (int r = 0; r < 8; ++r) {
        const int grow = bM + wm * 32 + mt * 16 + r + 8 * hi;  // C layout row
        const int gcol = bN + wn * 64 + nt * 16 + ln;          // C layout col
        float val = acc[mt][nt][r] + bias[gcol];
        if (GELU) val = 0.5f * val * (1.f + erff(val * 0.70710678118f));
        const size_t idx = (size_t)grow * N + gcol;
        if (RES) outF[idx] = val + residual[idx];
        else     outH[idx] = (half_t)val;
      }
}

// ---------------------------------------------------------------------------
// Flash attention (causal). One WG = 8 waves = 128 query rows of one (b,h).
// Each wave owns a 16-row query stripe; streams 32-key blocks:
//   S = Q(16x64) K^T  -> online softmax -> O += P(16x32) V(32x64)
// K^T B-fragments are 32B-aligned v16h global loads; V is staged transposed
// in LDS (reads = aligned v8h); P transposes C->A layout via per-wave LDS.
// ---------------------------------------------------------------------------
__global__ void attn_flash(const half_t* __restrict__ q, const half_t* __restrict__ k,
                           const half_t* __restrict__ v, half_t* __restrict__ ctx) {
  __shared__ __attribute__((aligned(16))) half_t Vt[64 * LDP];     // [d][key]
  __shared__ __attribute__((aligned(16))) half_t Pl[8][16 * LDP];  // per-wave [row][key]

  const int tid  = threadIdx.x;
  const int lane = tid & 31, wave = tid >> 5;
  const int hi = lane >> 4, ln = lane & 15;
  const int qt = blockIdx.x;
  const int b = blockIdx.y >> 4, h = blockIdx.y & 15;
  const size_t tokbase = (size_t)b * SEQ;
  const size_t hoff = (size_t)h * HEAD_DIM;
  const int q0 = qt * 128 + wave * 16;           // first seq position of stripe

  // Q A-fragments (held in registers for the whole key loop)
  v16h aq[2];
#pragma unroll
  for (int ds = 0; ds < 2; ++ds) {
    const half_t* qp = q + (tokbase + q0 + ln) * EMBED + hoff + ds * 32;
    aq[ds] = cat8(*(const v8h*)(qp + 16 * hi), *(const v8h*)(qp + 16 * hi + 8));
  }

  float m_i[8], l_i[8];
  v8f o[4];
#pragma unroll
  for (int r = 0; r < 8; r++) { m_i[r] = -1e30f; l_i[r] = 0.f; }
#pragma unroll
  for (int t = 0; t < 4; t++)
#pragma unroll
    for (int e = 0; e < 8; e++) o[t][e] = 0.f;

  const int kbmax = qt * 4 + 3;                  // WG-wide bound (keys <= qt*128+127)
  for (int kb = 0; kb <= kbmax; ++kb) {
    const int kbase = kb * 32;
    __syncthreads();
    {   // cooperative V stage, transposed: Vt[d][key]
      const int key = tid >> 3, d0 = (tid & 7) * 8;
      const half_t* vp = v + (tokbase + kbase + key) * EMBED + hoff + d0;
#pragma unroll
      for (int j = 0; j < 8; j++) Vt[(d0 + j) * LDP + key] = vp[j];
      if (kb < kbmax)
        __builtin_prefetch(v + (tokbase + kbase + 32 + key) * EMBED + hoff + d0, 0, 1);
    }
    __syncthreads();

    if (kbase <= q0 + 15) {                      // this wave still has unmasked keys
      v8f s0, s1;
#pragma unroll
      for (int e = 0; e < 8; e++) { s0[e] = 0.f; s1[e] = 0.f; }
#pragma unroll
      for (int ds = 0; ds < 2; ++ds) {
        // B[k=d][n=key] = K[key][d]: per-lane 16 contiguous halves (32B aligned)
        const half_t* kp0 = k + (tokbase + kbase + ln) * EMBED + hoff + ds * 32 + 16 * hi;
        v16h bk0 = *(const v16h*)(kp0);
        v16h bk1 = *(const v16h*)(kp0 + 16 * EMBED);
        s0 = wmma_f16(aq[ds], bk0, s0);
        s1 = wmma_f16(aq[ds], bk1, s1);
      }

      float a0[8], a1[8], rm[8], rs[8], alpha[8];
#pragma unroll
      for (int r = 0; r < 8; r++) {
        const int qs = q0 + r + 8 * hi;
        a0[r] = (kbase + ln      <= qs) ? s0[r] * 0.125f : -1e30f;  // 1/sqrt(64)
        a1[r] = (kbase + 16 + ln <= qs) ? s1[r] * 0.125f : -1e30f;
        rm[r] = fmaxf(a0[r], a1[r]);
      }
#pragma unroll
      for (int msk = 1; msk < 16; msk <<= 1)
#pragma unroll
        for (int r = 0; r < 8; r++) rm[r] = fmaxf(rm[r], __shfl_xor(rm[r], msk, 32));
#pragma unroll
      for (int r = 0; r < 8; r++) {
        const float mnew = fmaxf(m_i[r], rm[r]);
        alpha[r] = __expf(m_i[r] - mnew);
        m_i[r] = mnew;
        a0[r] = __expf(a0[r] - mnew);
        a1[r] = __expf(a1[r] - mnew);
        rs[r] = a0[r] + a1[r];
      }
#pragma unroll
      for (int msk = 1; msk < 16; msk <<= 1)
#pragma unroll
        for (int r = 0; r < 8; r++) rs[r] += __shfl_xor(rs[r], msk, 32);
#pragma unroll
      for (int r = 0; r < 8; r++) l_i[r] = l_i[r] * alpha[r] + rs[r];
#pragma unroll
      for (int t = 0; t < 4; t++)
#pragma unroll
        for (int r = 0; r < 8; r++) o[t][r] *= alpha[r];

      // P: C layout -> A layout via per-wave LDS (wave-private, DScnt fence only)
#pragma unroll
      for (int r = 0; r < 8; r++) {
        Pl[wave][(r + 8 * hi) * LDP + ln]      = (half_t)a0[r];
        Pl[wave][(r + 8 * hi) * LDP + 16 + ln] = (half_t)a1[r];
      }
      wave_lds_fence();
      const half_t* pb = &Pl[wave][ln * LDP];
      v16h ap = cat8(*(const v8h*)(pb + 16 * hi), *(const v8h*)(pb + 16 * hi + 8));

#pragma unroll
      for (int ot = 0; ot < 4; ++ot) {
        const half_t* vb = Vt + (ot * 16 + ln) * LDP + 16 * hi;
        v16h bvv = cat8(*(const v8h*)(vb), *(const v8h*)(vb + 8));
        o[ot] = wmma_f16(ap, bvv, o[ot]);
      }
    }
  }

#pragma unroll
  for (int ot = 0; ot < 4; ++ot)
#pragma unroll
    for (int r = 0; r < 8; r++) {
      const int qs = q0 + r + 8 * hi;
      const float val = o[ot][r] / l_i[r];
      ctx[(tokbase + qs) * EMBED + hoff + ot * 16 + ln] = (half_t)val;
    }
}

// ---------------------------------------------------------------------------
// Host-side pipeline
// ---------------------------------------------------------------------------
extern "C" void kernel_launch(void* const* d_in, const int* in_sizes, int n_in,
                              void* d_out, int out_size, void* d_ws, size_t ws_size,
                              hipStream_t stream) {
  const float* x    = (const float*)d_in[0];
  const float* ln1s = (const float*)d_in[1];
  const float* ln1b = (const float*)d_in[2];
  const float* ln2s = (const float*)d_in[3];
  const float* ln2b = (const float*)d_in[4];
  const float* wq   = (const float*)d_in[5];
  const float* bq   = (const float*)d_in[6];
  const float* wk   = (const float*)d_in[7];
  const float* bk   = (const float*)d_in[8];
  const float* wv   = (const float*)d_in[9];
  const float* bv   = (const float*)d_in[10];
  const float* wo   = (const float*)d_in[11];
  const float* bo   = (const float*)d_in[12];
  const float* w1   = (const float*)d_in[13];
  const float* b1   = (const float*)d_in[14];
  const float* w2   = (const float*)d_in[15];
  const float* b2   = (const float*)d_in[16];
  float* out = (float*)d_out;

  char* p = (char*)d_ws;
  auto alloc = [&](size_t bytes) -> char* {
    char* r = p;
    p += (bytes + 255) & ~(size_t)255;
    return r;
  };
  half_t* wq16 = (half_t*)alloc((size_t)EMBED * EMBED * 2);
  half_t* wk16 = (half_t*)alloc((size_t)EMBED * EMBED * 2);
  half_t* wv16 = (half_t*)alloc((size_t)EMBED * EMBED * 2);
  half_t* wo16 = (half_t*)alloc((size_t)EMBED * EMBED * 2);
  half_t* w116 = (half_t*)alloc((size_t)EMBED * HIDDEN * 2);
  half_t* w216 = (half_t*)alloc((size_t)HIDDEN * EMBED * 2);
  half_t* h16  = (half_t*)alloc((size_t)TOKENS * EMBED * 2);
  half_t* q16  = (half_t*)alloc((size_t)TOKENS * EMBED * 2);
  half_t* k16  = (half_t*)alloc((size_t)TOKENS * EMBED * 2);
  half_t* v16  = (half_t*)alloc((size_t)TOKENS * EMBED * 2);
  half_t* c16  = (half_t*)alloc((size_t)TOKENS * EMBED * 2);
  float*  x1   = (float*)alloc((size_t)TOKENS * EMBED * 4);
  half_t* H16  = (half_t*)alloc((size_t)TOKENS * HIDDEN * 2);

  // 1) weights fp32 -> f16
  cvt_f32_to_f16<<<512, 256, 0, stream>>>(wq, wq16, EMBED * EMBED);
  cvt_f32_to_f16<<<512, 256, 0, stream>>>(wk, wk16, EMBED * EMBED);
  cvt_f32_to_f16<<<512, 256, 0, stream>>>(wv, wv16, EMBED * EMBED);
  cvt_f32_to_f16<<<512, 256, 0, stream>>>(wo, wo16, EMBED * EMBED);
  cvt_f32_to_f16<<<1024, 256, 0, stream>>>(w1, w116, EMBED * HIDDEN);
  cvt_f32_to_f16<<<1024, 256, 0, stream>>>(w2, w216, HIDDEN * EMBED);

  // 2) LN1
  layernorm_f16<<<TOKENS, 256, 0, stream>>>(x, ln1s, ln1b, h16);

  // 3) Q/K/V projections
  dim3 gQKV(EMBED / 128, TOKENS / 128);
  gemm_wmma<false, false><<<gQKV, 256, 0, stream>>>(h16, wq16, bq, nullptr, nullptr, q16,
                                                    TOKENS, EMBED, EMBED);
  gemm_wmma<false, false><<<gQKV, 256, 0, stream>>>(h16, wk16, bk, nullptr, nullptr, k16,
                                                    TOKENS, EMBED, EMBED);
  gemm_wmma<false, false><<<gQKV, 256, 0, stream>>>(h16, wv16, bv, nullptr, nullptr, v16,
                                                    TOKENS, EMBED, EMBED);

  // 4) causal flash attention
  attn_flash<<<dim3(SEQ / 128, BATCH * HEADS), 256, 0, stream>>>(q16, k16, v16, c16);

  // 5) output projection + residual -> x1 (f32)
  gemm_wmma<false, true><<<gQKV, 256, 0, stream>>>(c16, wo16, bo, x, x1, nullptr,
                                                   TOKENS, EMBED, EMBED);

  // 6) LN2
  layernorm_f16<<<TOKENS, 256, 0, stream>>>(x1, ln2s, ln2b, h16);

  // 7) MLP up + exact GELU
  dim3 gUP(HIDDEN / 128, TOKENS / 128);
  gemm_wmma<true, false><<<gUP, 256, 0, stream>>>(h16, w116, b1, nullptr, nullptr, H16,
                                                  TOKENS, HIDDEN, EMBED);

  // 8) MLP down + residual -> out (f32)
  gemm_wmma<false, true><<<gQKV, 256, 0, stream>>>(H16, w216, b2, x1, out, nullptr,
                                                   TOKENS, EMBED, HIDDEN);
}
